// KGPrompt_45698452030124
// MI455X (gfx1250) — compile-verified, matrix operand order checked by reading, not run on hardware
//
#include <hip/hip_runtime.h>
#include <hip/hip_bf16.h>

typedef __attribute__((ext_vector_type(16))) _Float16 v16h;
typedef __attribute__((ext_vector_type(8)))  _Float16 v8h;
typedef __attribute__((ext_vector_type(8)))  float    v8f;
typedef __attribute__((ext_vector_type(4)))  int      v4i;

// ---- problem constants (match reference) ----
#define N_ENTITY 20000
#define N_EDGE   200000
#define N_REL    12
#define N_BASES  8
#define DD       384
#define HH       768
#define BB       32
#define LE       32
#define LT       64
#define NPFX     10
#define LL       74          // NPFX + LT
#define M_ENT    1024        // B*LE
#define M_TOK    2048        // B*LT
#define M_PR     2368        // B*LL
#define NOUT     18432       // 12*2*768

// CDNA5 async global->LDS path (ASYNCcnt), with safe fallback.
#if defined(__has_builtin)
#  if __has_builtin(__builtin_amdgcn_global_load_async_to_lds_b128) && \
      __has_builtin(__builtin_amdgcn_s_wait_asynccnt)
#    define HAS_ASYNC 1
#  endif
#endif
#ifndef HAS_ASYNC
#  define HAS_ASYNC 0
#endif

typedef __attribute__((address_space(1))) v4i glob_v4i;
typedef __attribute__((address_space(3))) v4i lds_v4i;

__device__ __forceinline__ int iminD(int a, int b) { return a < b ? a : b; }

__device__ __forceinline__ void copy16_to_lds(const _Float16* g, _Float16* l) {
#if HAS_ASYNC
  __builtin_amdgcn_global_load_async_to_lds_b128(
      (glob_v4i*)g, (lds_v4i*)l, 0, 0);
#else
  *(v8h*)l = *(const v8h*)g;
#endif
}

// Fragment read from LDS for v_wmma_f32_16x16x32_f16:
// elements 0..7 = K=k0..k0+7, elements 8..15 = K=k0+16..k0+23 (per-lane).
__device__ __forceinline__ v16h frag_lds(const _Float16* p) {
  const v8h lo = *(const v8h*)(p);
  const v8h hi = *(const v8h*)(p + 16);
  return __builtin_shufflevector(lo, hi, 0,1,2,3,4,5,6,7,8,9,10,11,12,13,14,15);
}

#define APITCH 72   // 64 + 8 halves: 144B row pitch -> conflict-free ds_load_b128
#define BPITCH 72

// ---------------- WMMA GEMM with async-LDS double buffering ----------------
// C[M,N] = A[M,K](f16,row-major) x Bt[N,K](f16, N-major == B col-major)
// Block tile 64x128, 8 waves (4 row-groups x 2 col-groups), wave tile 16x64.
// K-tile 64 (two WMMA k-steps per stage). flags: 1 = relu, 2 = C32 += result.
__global__ __launch_bounds__(256)
void k_gemm(const _Float16* __restrict__ A, const _Float16* __restrict__ Bt,
            const float* __restrict__ bias, const float* __restrict__ resid,
            float* __restrict__ C32, _Float16* __restrict__ C16,
            int M, int N, int K, int flags) {
  __shared__ _Float16 As[2][64 * APITCH];
  __shared__ _Float16 Bs[2][128 * BPITCH];
  const int tid   = threadIdx.x;
  const int lane  = tid & 31;
  const int wave  = tid >> 5;
  const int wr    = wave >> 1;       // 0..3
  const int wc    = wave & 1;        // 0..1
  const int m0    = blockIdx.y * 64 + wr * 16;
  const int n0    = blockIdx.x * 128 + wc * 64;
  const int col16 = lane & 15;
  const int kb    = (lane < 16) ? 0 : 8;

  auto load_stage = [&](int buf, int k0) {
#pragma unroll
    for (int j = 0; j < 2; ++j) {            // A: 64 rows x 8 chunks of 8 halves
      int idx = tid + j * 256;
      int r = idx >> 3, c = idx & 7;
      int gr = iminD((int)blockIdx.y * 64 + r, M - 1);
      copy16_to_lds(A + (size_t)gr * K + k0 + c * 8, &As[buf][r * APITCH + c * 8]);
    }
#pragma unroll
    for (int j = 0; j < 4; ++j) {            // B: 128 rows x 8 chunks
      int idx = tid + j * 256;
      int r = idx >> 3, c = idx & 7;
      int gr = iminD((int)blockIdx.x * 128 + r, N - 1);
      copy16_to_lds(Bt + (size_t)gr * K + k0 + c * 8, &Bs[buf][r * BPITCH + c * 8]);
    }
  };

  v8f acc[4] = {};
  load_stage(0, 0);
  const int S = K >> 6;
  for (int s = 0; s < S; ++s) {
    const int buf = s & 1;
    if (s + 1 < S) {
      load_stage(buf ^ 1, (s + 1) << 6);
#if HAS_ASYNC
      __builtin_amdgcn_s_wait_asynccnt(6);   // stage s complete; s+1 in flight
#endif
    } else {
#if HAS_ASYNC
      __builtin_amdgcn_s_wait_asynccnt(0);
#endif
    }
    __syncthreads();                         // stage s visible to all waves
#pragma unroll
    for (int kk = 0; kk < 2; ++kk) {
      v16h a = frag_lds(&As[buf][(wr * 16 + col16) * APITCH + kk * 32 + kb]);
#pragma unroll
      for (int st = 0; st < 4; ++st) {
        v16h b = frag_lds(&Bs[buf][(wc * 64 + st * 16 + col16) * BPITCH + kk * 32 + kb]);
        acc[st] = __builtin_amdgcn_wmma_f32_16x16x32_f16(
            false, a, false, b, (short)0, acc[st], false, false);
      }
    }
    __syncthreads();                         // done reading buf before s+2 overwrites
  }

  const int rbase = m0 + ((lane >= 16) ? 8 : 0);
#pragma unroll
  for (int st = 0; st < 4; ++st) {
    int col = n0 + st * 16 + col16;
    if (col >= N) continue;
    float bv = bias ? bias[col] : 0.0f;
#pragma unroll
    for (int i = 0; i < 8; ++i) {
      int row = rbase + i;
      if (row >= M) continue;
      float v = acc[st][i] + bv;
      if (flags & 1) v = v > 0.0f ? v : 0.0f;
      size_t idx = (size_t)row * N + col;
      if (resid) v += resid[idx];
      if (flags & 2) v += C32[idx];
      if (C32) C32[idx] = v;
      if (C16) C16[idx] = (_Float16)v;
    }
  }
}

// ---------------- final GEMM with fused 6-D transpose epilogue ----------------
// OUT = P2[M_PR,768] x pp2t[NOUT,768] + pp2_b into d_out[12,2,32,12,74,64].
__global__ __launch_bounds__(256)
void k_gemm_out(const _Float16* __restrict__ A, const _Float16* __restrict__ Bt,
                const float* __restrict__ bias, float* __restrict__ out) {
  const int M = M_PR, N = NOUT, K = HH;
  __shared__ _Float16 As[2][64 * APITCH];
  __shared__ _Float16 Bs[2][128 * BPITCH];
  const int tid   = threadIdx.x;
  const int lane  = tid & 31;
  const int wave  = tid >> 5;
  const int wr    = wave >> 1;
  const int wc    = wave & 1;
  const int m0    = blockIdx.y * 64 + wr * 16;
  const int n0    = blockIdx.x * 128 + wc * 64;
  const int col16 = lane & 15;
  const int kb    = (lane < 16) ? 0 : 8;

  auto load_stage = [&](int buf, int k0) {
#pragma unroll
    for (int j = 0; j < 2; ++j) {
      int idx = tid + j * 256;
      int r = idx >> 3, c = idx & 7;
      int gr = iminD((int)blockIdx.y * 64 + r, M - 1);
      copy16_to_lds(A + (size_t)gr * K + k0 + c * 8, &As[buf][r * APITCH + c * 8]);
    }
#pragma unroll
    for (int j = 0; j < 4; ++j) {
      int idx = tid + j * 256;
      int r = idx >> 3, c = idx & 7;
      int gr = iminD((int)blockIdx.x * 128 + r, N - 1);
      copy16_to_lds(Bt + (size_t)gr * K + k0 + c * 8, &Bs[buf][r * BPITCH + c * 8]);
    }
  };

  v8f acc[4] = {};
  load_stage(0, 0);
  const int S = K >> 6;
  for (int s = 0; s < S; ++s) {
    const int buf = s & 1;
    if (s + 1 < S) {
      load_stage(buf ^ 1, (s + 1) << 6);
#if HAS_ASYNC
      __builtin_amdgcn_s_wait_asynccnt(6);
#endif
    } else {
#if HAS_ASYNC
      __builtin_amdgcn_s_wait_asynccnt(0);
#endif
    }
    __syncthreads();
#pragma unroll
    for (int kk = 0; kk < 2; ++kk) {
      v16h a = frag_lds(&As[buf][(wr * 16 + col16) * APITCH + kk * 32 + kb]);
#pragma unroll
      for (int st = 0; st < 4; ++st) {
        v16h b = frag_lds(&Bs[buf][(wc * 64 + st * 16 + col16) * BPITCH + kk * 32 + kb]);
        acc[st] = __builtin_amdgcn_wmma_f32_16x16x32_f16(
            false, a, false, b, (short)0, acc[st], false, false);
      }
    }
    __syncthreads();
  }

  const int rbase = m0 + ((lane >= 16) ? 8 : 0);
#pragma unroll
  for (int st = 0; st < 4; ++st) {
    int col = n0 + st * 16 + col16;
    float bv = bias[col];
    int l   = col / 1536;
    int blk = (col % 1536) / 768;
    int hh  = (col % 768) / 64;
    int d   = col & 63;
#pragma unroll
    for (int i = 0; i < 8; ++i) {
      int row = rbase + i;
      if (row >= M) continue;
      int b = row / LL, t = row % LL;
      size_t idx = (((((size_t)((l * 2 + blk) * BB + b)) * 12 + hh) * LL + t) * 64) + d;
      out[idx] = acc[st][i] + bv;
    }
  }
}

// ---------------- small utility kernels ----------------
__global__ void k_fill_f32(float* p, size_t n, float v) {
  size_t i = (size_t)blockIdx.x * blockDim.x + threadIdx.x;
  if (i < n) p[i] = v;
}
__global__ void k_fill_i32(int* p, size_t n, int v) {
  size_t i = (size_t)blockIdx.x * blockDim.x + threadIdx.x;
  if (i < n) p[i] = v;
}
__global__ void k_convert16(const float* __restrict__ x, _Float16* __restrict__ y, size_t n) {
  size_t i = (size_t)blockIdx.x * blockDim.x + threadIdx.x;
  if (i < n) y[i] = (_Float16)x[i];
}
// Wt[N,K] (f16) from W[K,N] (f32)
__global__ void k_transpose16(const float* __restrict__ W, _Float16* __restrict__ Wt,
                              int K, int N) {
  size_t i = (size_t)blockIdx.x * blockDim.x + threadIdx.x;
  if (i >= (size_t)K * N) return;
  int n = (int)(i / K), k = (int)(i % K);
  Wt[i] = (_Float16)W[(size_t)k * N + n];
}
// Wt_r[r][o][i] = sum_b comp[r,b]*basis[b,i,o]  (B operand for agg GEMM, N-major)
__global__ void k_wcomp(const float* __restrict__ comp, const float* __restrict__ basis,
                        _Float16* __restrict__ wtr) {
  size_t idx = (size_t)blockIdx.x * blockDim.x + threadIdx.x;
  if (idx >= (size_t)N_REL * DD * DD) return;
  int i = (int)(idx % DD);
  int o = (int)((idx / DD) % DD);
  int r = (int)(idx / ((size_t)DD * DD));
  float acc = 0.0f;
#pragma unroll
  for (int b = 0; b < N_BASES; ++b)
    acc += comp[r * N_BASES + b] * basis[((size_t)b * DD + i) * DD + o];
  wtr[idx] = (_Float16)acc;
}
__global__ void k_build_slot(const int* __restrict__ ids, int* slot) {
  int i = blockIdx.x * blockDim.x + threadIdx.x;
  if (i < M_ENT) atomicCAS(&slot[ids[i]], -1, i);
}
__global__ void k_gatherx(const int* __restrict__ ids, const float* __restrict__ node,
                          float* __restrict__ x32, _Float16* __restrict__ x16) {
  size_t i = (size_t)blockIdx.x * blockDim.x + threadIdx.x;
  if (i >= (size_t)M_ENT * DD) return;
  int row = (int)(i / DD), d = (int)(i % DD);
  float v = node[(size_t)ids[row] * DD + d];
  x32[i] = v; x16[i] = (_Float16)v;
}
__global__ void k_cnt(const int* __restrict__ dst, const int* __restrict__ type,
                      const int* __restrict__ slot, float* __restrict__ cnt) {
  int e = blockIdx.x * blockDim.x + threadIdx.x;
  if (e >= N_EDGE) return;
  int s = slot[dst[e]];
  if (s >= 0) atomicAdd(&cnt[type[e] * M_ENT + s], 1.0f);
}
__global__ void k_scatter(const int* __restrict__ src, const int* __restrict__ dst,
                          const int* __restrict__ type, const int* __restrict__ slot,
                          const float* __restrict__ node, float* __restrict__ sums) {
  size_t idx = (size_t)blockIdx.x * blockDim.x + threadIdx.x;
  if (idx >= (size_t)N_EDGE * DD) return;
  int e = (int)(idx / DD), d = (int)(idx % DD);
  int s = slot[dst[e]];
  if (s >= 0)
    atomicAdd(&sums[((size_t)type[e] * M_ENT + s) * DD + d],
              node[(size_t)src[e] * DD + d]);
}
// G16[i,d] = (f16)(sums[r, slot(ids[i]), d] / max(cnt,1))
__global__ void k_scale_gather(const int* __restrict__ ids, const int* __restrict__ slot,
                               const float* __restrict__ sums, const float* __restrict__ cnt,
                               _Float16* __restrict__ G, int r) {
  size_t i = (size_t)blockIdx.x * blockDim.x + threadIdx.x;
  if (i >= (size_t)M_ENT * DD) return;
  int row = (int)(i / DD), d = (int)(i % DD);
  int s = slot[ids[row]];
  float c = fmaxf(cnt[r * M_ENT + s], 1.0f);
  G[i] = (_Float16)(sums[((size_t)r * M_ENT + s) * DD + d] / c);
}
__global__ void k_axpy(float* __restrict__ y, const float* __restrict__ x, size_t n) {
  size_t i = (size_t)blockIdx.x * blockDim.x + threadIdx.x;
  if (i < n) y[i] += x[i];
}
// cross-attention: scores over 32 entities, softmax, weighted sum + residual
__global__ __launch_bounds__(256)
void k_attn(const float* __restrict__ q, const float* __restrict__ entH,
            const float* __restrict__ tokH, float* __restrict__ prompt32,
            _Float16* __restrict__ prompt16) {
  int bt = blockIdx.x;                 // 0..2047
  int b = bt / LT, t = bt % LT;
  const float* qv = q + (size_t)bt * HH;
  __shared__ float sc[LE];
  __shared__ float red[256];
  for (int e = 0; e < LE; ++e) {
    const float* ev = entH + (size_t)(b * LE + e) * HH;
    float p = 0.0f;
    for (int h = threadIdx.x; h < HH; h += 256) p += qv[h] * ev[h];
    red[threadIdx.x] = p; __syncthreads();
    for (int s = 128; s > 0; s >>= 1) {
      if ((int)threadIdx.x < s) red[threadIdx.x] += red[threadIdx.x + s];
      __syncthreads();
    }
    if (threadIdx.x == 0) sc[e] = red[0] / (float)HH;
    __syncthreads();
  }
  if (threadIdx.x == 0) {
    float m = sc[0];
    for (int e = 1; e < LE; ++e) m = fmaxf(m, sc[e]);
    float s = 0.0f;
    for (int e = 0; e < LE; ++e) { sc[e] = expf(sc[e] - m); s += sc[e]; }
    float inv = 1.0f / s;
    for (int e = 0; e < LE; ++e) sc[e] *= inv;
  }
  __syncthreads();
  for (int h = threadIdx.x; h < HH; h += 256) {
    float acc = tokH[(size_t)bt * HH + h];
    for (int e = 0; e < LE; ++e) acc += sc[e] * entH[(size_t)(b * LE + e) * HH + h];
    size_t row = (size_t)b * LL + NPFX + t;
    prompt32[row * HH + h] = acc;
    prompt16[row * HH + h] = (_Float16)acc;
  }
}
__global__ void k_prefix_bcast(const float* __restrict__ pre32,
                               float* __restrict__ prompt32,
                               _Float16* __restrict__ prompt16) {
  size_t i = (size_t)blockIdx.x * blockDim.x + threadIdx.x;
  if (i >= (size_t)BB * NPFX * HH) return;
  int b = (int)(i / (NPFX * HH));
  int r = (int)((i / HH) % NPFX);
  int h = (int)(i % HH);
  float v = pre32[(size_t)r * HH + h];
  size_t row = (size_t)b * LL + r;
  prompt32[row * HH + h] = v;
  prompt16[row * HH + h] = (_Float16)v;
}

// ---------------- host ----------------
extern "C" void kernel_launch(void* const* d_in, const int* in_sizes, int n_in,
                              void* d_out, int out_size, void* d_ws, size_t ws_size,
                              hipStream_t stream) {
  const int*   entity_ids = (const int*)  d_in[0];
  const float* token_emb  = (const float*)d_in[1];
  const int*   edge_index = (const int*)  d_in[2];
  const int*   edge_type  = (const int*)  d_in[3];
  const float* node_emb   = (const float*)d_in[4];
  const float* basis      = (const float*)d_in[5];
  const float* comp       = (const float*)d_in[6];
  const float* root       = (const float*)d_in[7];
  const float* rgcn_bias  = (const float*)d_in[8];
  const float* ep1_w1 = (const float*)d_in[9];  const float* ep1_b1 = (const float*)d_in[10];
  const float* ep1_w2 = (const float*)d_in[11]; const float* ep1_b2 = (const float*)d_in[12];
  const float* ep2_w  = (const float*)d_in[13]; const float* ep2_b  = (const float*)d_in[14];
  const float* tp1_w1 = (const float*)d_in[15]; const float* tp1_b1 = (const float*)d_in[16];
  const float* tp1_w2 = (const float*)d_in[17]; const float* tp1_b2 = (const float*)d_in[18];
  const float* tp2_w  = (const float*)d_in[19]; const float* tp2_b  = (const float*)d_in[20];
  const float* xattn_w = (const float*)d_in[21];
  const float* pp1_w1 = (const float*)d_in[22]; const float* pp1_b1 = (const float*)d_in[23];
  const float* pp1_w2 = (const float*)d_in[24]; const float* pp1_b2 = (const float*)d_in[25];
  const float* pp2_w  = (const float*)d_in[26]; const float* pp2_b  = (const float*)d_in[27];
  const float* rec_emb = (const float*)d_in[28];
  const float* rp_w1 = (const float*)d_in[29]; const float* rp_b1 = (const float*)d_in[30];
  const float* rp_w2 = (const float*)d_in[31]; const float* rp_b2 = (const float*)d_in[32];

  size_t off = 0;
  auto alloc = [&](size_t bytes) -> void* {
    void* p = (char*)d_ws + off;
    off += (bytes + 255) & ~(size_t)255;
    return p;
  };

  _Float16* wtr     = (_Float16*)alloc((size_t)N_REL * DD * DD * 2);
  _Float16* root_t  = (_Float16*)alloc((size_t)DD * DD * 2);
  _Float16* ep1w1_t = (_Float16*)alloc((size_t)192 * DD * 2);
  _Float16* ep1w2_t = (_Float16*)alloc((size_t)DD * 192 * 2);
  _Float16* ep2w_t  = (_Float16*)alloc((size_t)HH * DD * 2);
  _Float16* tp1w1_t = (_Float16*)alloc((size_t)DD * HH * 2);
  _Float16* tp1w2_t = (_Float16*)alloc((size_t)HH * DD * 2);
  _Float16* tp2w_t  = (_Float16*)alloc((size_t)HH * HH * 2);
  _Float16* xatt_t  = (_Float16*)alloc((size_t)HH * HH * 2);
  _Float16* pp1w1_t = (_Float16*)alloc((size_t)DD * HH * 2);
  _Float16* pp1w2_t = (_Float16*)alloc((size_t)HH * DD * 2);
  _Float16* pp2w_t  = (_Float16*)alloc((size_t)NOUT * HH * 2);
  _Float16* rpw1_t  = (_Float16*)alloc((size_t)DD * HH * 2);
  _Float16* rpw2_t  = (_Float16*)alloc((size_t)HH * DD * 2);
  int*   slot   = (int*)  alloc((size_t)N_ENTITY * 4);
  float* cnt_s  = (float*)alloc((size_t)N_REL * M_ENT * 4);
  float* sums_s = (float*)alloc((size_t)N_REL * M_ENT * DD * 4);
  _Float16* xg16   = (_Float16*)alloc((size_t)M_ENT * DD * 2);
  float*    xg32   = (float*)   alloc((size_t)M_ENT * DD * 4);
  _Float16* G16    = (_Float16*)alloc((size_t)M_ENT * DD * 2);
  float*    agg32  = (float*)   alloc((size_t)M_ENT * DD * 4);
  _Float16* ent16  = (_Float16*)alloc((size_t)M_ENT * DD * 2);
  float*    ent32  = (float*)   alloc((size_t)M_ENT * DD * 4);
  _Float16* eh16   = (_Float16*)alloc((size_t)M_ENT * 192 * 2);
  _Float16* ent2_16= (_Float16*)alloc((size_t)M_ENT * DD * 2);
  float*    entH32 = (float*)   alloc((size_t)M_ENT * HH * 4);
  _Float16* tok16   = (_Float16*)alloc((size_t)M_TOK * HH * 2);
  _Float16* th16    = (_Float16*)alloc((size_t)M_TOK * DD * 2);
  _Float16* tok2_16 = (_Float16*)alloc((size_t)M_TOK * HH * 2);
  float*    tokH32  = (float*)   alloc((size_t)M_TOK * HH * 4);
  _Float16* tokH16  = (_Float16*)alloc((size_t)M_TOK * HH * 2);
  float*    q32     = (float*)   alloc((size_t)M_TOK * HH * 4);
  _Float16* rec16  = (_Float16*)alloc((size_t)NPFX * HH * 2);
  _Float16* ph16   = (_Float16*)alloc((size_t)NPFX * DD * 2);
  float*    pre32  = (float*)   alloc((size_t)NPFX * HH * 4);
  float*    prompt32 = (float*)   alloc((size_t)M_PR * HH * 4);
  _Float16* prompt16 = (_Float16*)alloc((size_t)M_PR * HH * 2);
  _Float16* pm16   = (_Float16*)alloc((size_t)M_PR * DD * 2);
  _Float16* p2_16  = (_Float16*)alloc((size_t)M_PR * HH * 2);

  auto blocks = [](size_t n) { return (unsigned)((n + 255) / 256); };
  auto tr = [&](const float* W, _Float16* Wt, int K, int N) {
    k_transpose16<<<blocks((size_t)K * N), 256, 0, stream>>>(W, Wt, K, N);
  };
  auto gemm = [&](const _Float16* A, const _Float16* Bt, const float* bias,
                  const float* resid, float* C32, _Float16* C16,
                  int M, int N, int K, int flags) {
    dim3 g((N + 127) / 128, (M + 63) / 64);
    k_gemm<<<g, 256, 0, stream>>>(A, Bt, bias, resid, C32, C16, M, N, K, flags);
  };

  // --- weight prep (f32 -> f16, N-major for WMMA B operand) ---
  k_wcomp<<<blocks((size_t)N_REL * DD * DD), 256, 0, stream>>>(comp, basis, wtr);
  tr(root,   root_t,  DD, DD);
  tr(ep1_w1, ep1w1_t, DD, 192);  tr(ep1_w2, ep1w2_t, 192, DD);
  tr(ep2_w,  ep2w_t,  DD, HH);
  tr(tp1_w1, tp1w1_t, HH, DD);   tr(tp1_w2, tp1w2_t, DD, HH);
  tr(tp2_w,  tp2w_t,  HH, HH);   tr(xattn_w, xatt_t, HH, HH);
  tr(pp1_w1, pp1w1_t, HH, DD);   tr(pp1_w2, pp1w2_t, DD, HH);
  tr(pp2_w,  pp2w_t,  HH, NOUT);
  tr(rp_w1,  rpw1_t,  HH, DD);   tr(rp_w2,  rpw2_t,  DD, HH);

  // --- RGCN: compact segment sums for the 1024 gathered nodes only ---
  k_fill_i32<<<blocks(N_ENTITY), 256, 0, stream>>>(slot, N_ENTITY, -1);
  k_fill_f32<<<blocks((size_t)N_REL * M_ENT), 256, 0, stream>>>(cnt_s, (size_t)N_REL * M_ENT, 0.f);
  k_fill_f32<<<blocks((size_t)N_REL * M_ENT * DD), 256, 0, stream>>>(sums_s, (size_t)N_REL * M_ENT * DD, 0.f);
  k_fill_f32<<<blocks((size_t)M_ENT * DD), 256, 0, stream>>>(agg32, (size_t)M_ENT * DD, 0.f);
  k_build_slot<<<blocks(M_ENT), 256, 0, stream>>>(entity_ids, slot);
  k_gatherx<<<blocks((size_t)M_ENT * DD), 256, 0, stream>>>(entity_ids, node_emb, xg32, xg16);
  const int* e_src = edge_index;
  const int* e_dst = edge_index + N_EDGE;
  k_cnt<<<blocks(N_EDGE), 256, 0, stream>>>(e_dst, edge_type, slot, cnt_s);
  k_scatter<<<blocks((size_t)N_EDGE * DD), 256, 0, stream>>>(e_src, e_dst, edge_type, slot, node_emb, sums_s);

  // agg += mean_r @ W_r   (12 WMMA GEMMs accumulating into agg32)
  for (int r = 0; r < N_REL; ++r) {
    k_scale_gather<<<blocks((size_t)M_ENT * DD), 256, 0, stream>>>(entity_ids, slot, sums_s, cnt_s, G16, r);
    gemm(G16, wtr + (size_t)r * DD * DD, nullptr, nullptr, agg32, nullptr, M_ENT, DD, DD, 2);
  }
  // ent = agg + x@root + bias + x
  k_axpy<<<blocks((size_t)M_ENT * DD), 256, 0, stream>>>(agg32, xg32, (size_t)M_ENT * DD);
  gemm(xg16, root_t, rgcn_bias, agg32, ent32, ent16, M_ENT, DD, DD, 0);
  // entity MLP + proj2
  gemm(ent16, ep1w1_t, ep1_b1, nullptr, nullptr, eh16, M_ENT, 192, DD, 1);
  gemm(eh16, ep1w2_t, ep1_b2, ent32, nullptr, ent2_16, M_ENT, DD, 192, 0);
  gemm(ent2_16, ep2w_t, ep2_b, nullptr, entH32, nullptr, M_ENT, HH, DD, 0);

  // token MLP + proj2 + query
  k_convert16<<<blocks((size_t)M_TOK * HH), 256, 0, stream>>>(token_emb, tok16, (size_t)M_TOK * HH);
  gemm(tok16, tp1w1_t, tp1_b1, nullptr, nullptr, th16, M_TOK, DD, HH, 1);
  gemm(th16, tp1w2_t, tp1_b2, token_emb, nullptr, tok2_16, M_TOK, HH, DD, 0);
  gemm(tok2_16, tp2w_t, tp2_b, nullptr, tokH32, tokH16, M_TOK, HH, HH, 0);
  gemm(tokH16, xatt_t, nullptr, nullptr, q32, nullptr, M_TOK, HH, HH, 0);

  // cross attention -> prompt rows [NPFX..LL)
  k_attn<<<M_TOK, 256, 0, stream>>>(q32, entH32, tokH32, prompt32, prompt16);

  // rec prefix MLP -> prompt rows [0..NPFX)
  k_convert16<<<blocks((size_t)NPFX * HH), 256, 0, stream>>>(rec_emb, rec16, (size_t)NPFX * HH);
  gemm(rec16, rpw1_t, rp_b1, nullptr, nullptr, ph16, NPFX, DD, HH, 1);
  gemm(ph16, rpw2_t, rp_b2, rec_emb, pre32, nullptr, NPFX, HH, DD, 0);
  k_prefix_bcast<<<blocks((size_t)BB * NPFX * HH), 256, 0, stream>>>(pre32, prompt32, prompt16);

  // prompt MLP
  gemm(prompt16, pp1w1_t, pp1_b1, nullptr, nullptr, pm16, M_PR, DD, HH, 1);
  gemm(pm16, pp1w2_t, pp1_b2, prompt32, nullptr, p2_16, M_PR, HH, DD, 0);

  // final 768 -> 18432 projection with fused [12,2,32,12,74,64] transpose
  {
    dim3 g((NOUT + 127) / 128, (M_PR + 63) / 64);
    k_gemm_out<<<g, 256, 0, stream>>>(p2_16, pp2w_t, pp2_b, (float*)d_out);
  }
}
